// SlidingWindowAttention_60808146977414
// MI455X (gfx1250) — matrix operand order repaired and checked by reading, not compiled, not run on hardware
//
#include <hip/hip_runtime.h>

typedef __bf16 bf16_t;
typedef __attribute__((ext_vector_type(16))) __bf16 v16bf;
typedef __attribute__((ext_vector_type(8)))  __bf16 v8bf;
typedef __attribute__((ext_vector_type(8)))  float  v8f;
typedef __attribute__((ext_vector_type(4)))  float  v4f;

#define SEQ_L    4096
#define DHEAD    128
#define WINDOW   2048
#define NSINK    4
#define BLOCK_M  64
#define BLOCK_N  32
#define THREADS  128      // 4 waves (wave32)

// padded LDS strides (bf16 elements); keep 16B alignment for b128 reads
#define KSTRIDE  (DHEAD + 8)     // 272 B/row
#define VSTRIDE  (BLOCK_N + 8)   // 80 B/row
#define PSTRIDE  (BLOCK_N + 8)   // 80 B/row

static __device__ __forceinline__ v16bf cat8(v8bf a, v8bf b) {
  return __builtin_shufflevector(a, b, 0,1,2,3,4,5,6,7,8,9,10,11,12,13,14,15);
}
// B-fragment: one contiguous 16-element run per lane
static __device__ __forceinline__ v16bf lds_load16(const bf16_t* p) {
  v8bf a = *(const v8bf*)p;
  v8bf b = *(const v8bf*)(p + 8);
  return cat8(a, b);
}
// A-fragment: two 8-element runs per lane
static __device__ __forceinline__ v16bf lds_load8x2(const bf16_t* p0, const bf16_t* p1) {
  v8bf a = *(const v8bf*)p0;
  v8bf b = *(const v8bf*)p1;
  return cat8(a, b);
}

__global__ __launch_bounds__(THREADS)
void swa_fwd_kernel(const float* __restrict__ Q, const float* __restrict__ K,
                    const float* __restrict__ V, float* __restrict__ O)
{
  __shared__ __align__(16) bf16_t Ksh[BLOCK_N * KSTRIDE];            // [key][d]
  __shared__ __align__(16) bf16_t VshT[DHEAD * VSTRIDE];             // [d][key] (transposed)
  __shared__ __align__(16) bf16_t Psh[(THREADS / 32) * 16 * PSTRIDE];// per-wave P staging

  const int tid  = threadIdx.x;
  const int wave = tid >> 5;
  const int lane = tid & 31;
  const int ln16 = lane & 15;
  const int hi   = lane >> 4;

  const int qs = blockIdx.x * BLOCK_M;
  const int bh = blockIdx.y;

  const float* Qb = Q + (size_t)bh * SEQ_L * DHEAD;
  const float* Kb = K + (size_t)bh * SEQ_L * DHEAD;
  const float* Vb = V + (size_t)bh * SEQ_L * DHEAD;
  float*       Ob = O + (size_t)bh * SEQ_L * DHEAD;

  const int qrow0 = qs + wave * 16;      // first query row of this wave
  const int qrow  = qrow0 + ln16;

  // ---- Q fragments in A-layout (one 16-elem K-run: 16*hi), fp32->bf16 ----
  v16bf qa[4];
#pragma unroll
  for (int c = 0; c < 4; ++c) {
    const float* p0 = Qb + (size_t)qrow * DHEAD + c * 32 + 16 * hi;
    const float* p1 = p0 + 8;
    v4f f0 = *(const v4f*)p0, f1 = *(const v4f*)(p0 + 4);
    v4f f2 = *(const v4f*)p1, f3 = *(const v4f*)(p1 + 4);
    v16bf t;
#pragma unroll
    for (int e = 0; e < 4; ++e) {
      t[e]      = (bf16_t)f0[e];
      t[4 + e]  = (bf16_t)f1[e];
      t[8 + e]  = (bf16_t)f2[e];
      t[12 + e] = (bf16_t)f3[e];
    }
    qa[c] = t;
  }

  v8f acc[8];
  const v8f vzero = {};
#pragma unroll
  for (int f = 0; f < 8; ++f) acc[f] = vzero;

  float mrow[8], lrow[8];
#pragma unroll
  for (int r = 0; r < 8; ++r) { mrow[r] = -1e30f; lrow[r] = 0.0f; }

  int klo = qs - (WINDOW - 1);
  if (klo < 0) klo = 0;
  const int kt0       = (klo / BLOCK_N) * BLOCK_N;
  const int klast     = qs + BLOCK_M - 1;                 // causal bound
  const int sink_tile = (kt0 > 0) ? 1 : 0;
  const int ntiles    = (klast - kt0) / BLOCK_N + 1 + sink_tile;

  // softmax done in exp2 domain: fold 1/sqrt(D) * log2(e) into the scores
  const float scale_log2 = 0.08838834764831845f * 1.4426950408889634f;

  for (int it = 0; it < ntiles; ++it) {
    const int kt = (sink_tile && it == 0) ? 0 : kt0 + (it - sink_tile) * BLOCK_N;

    // ---- cooperative K/V tile load, fp32 -> bf16, V transposed into LDS ----
#pragma unroll 8
    for (int i = 0; i < BLOCK_N; ++i) {
      float kv = Kb[(size_t)(kt + i) * DHEAD + tid];
      float vv = Vb[(size_t)(kt + i) * DHEAD + tid];
      Ksh[i * KSTRIDE + tid]  = (bf16_t)kv;
      VshT[tid * VSTRIDE + i] = (bf16_t)vv;
    }
    __syncthreads();

    // ---- S = Q K^T : 8x v_wmma_f32_16x16x32_bf16 ----
    v8f s0 = {}, s1 = {};
#pragma unroll
    for (int c = 0; c < 4; ++c) {
      const int d0 = c * 32 + 16 * hi;
      v16bf kb0 = lds_load16(&Ksh[ln16 * KSTRIDE + d0]);
      v16bf kb1 = lds_load16(&Ksh[(16 + ln16) * KSTRIDE + d0]);
      s0 = __builtin_amdgcn_wmma_f32_16x16x32_bf16(false, qa[c], false, kb0,
                                                   (short)0, s0, false, false);
      s1 = __builtin_amdgcn_wmma_f32_16x16x32_bf16(false, qa[c], false, kb1,
                                                   (short)0, s1, false, false);
    }

    // ---- mask (C-layout: VGPR r -> row r+8*hi, lane -> key col) ----
    float sv0[8], sv1[8];
#pragma unroll
    for (int r = 0; r < 8; ++r) { sv0[r] = s0[r] * scale_log2; sv1[r] = s1[r] * scale_log2; }

    const int n0 = kt + ln16;
    const int n1 = kt + 16 + ln16;
    const bool interior = (kt + BLOCK_N - 1 <= qrow0) &&
                          (kt >= qrow0 + 15 - (WINDOW - 1));
    if (!interior) {
#pragma unroll
      for (int r = 0; r < 8; ++r) {
        const int qp = qrow0 + r + 8 * hi;
        const bool a0 = (n0 <= qp) && ((n0 >= qp - (WINDOW - 1)) || (n0 < NSINK));
        const bool a1 = (n1 <= qp) && ((n1 >= qp - (WINDOW - 1)) || (n1 < NSINK));
        if (!a0) sv0[r] = -1e30f;
        if (!a1) sv1[r] = -1e30f;
      }
    }

    // ---- online softmax: row reductions across the 16-lane N groups ----
    float rm[8];
#pragma unroll
    for (int r = 0; r < 8; ++r) rm[r] = fmaxf(sv0[r], sv1[r]);
#pragma unroll
    for (int m = 1; m < 16; m <<= 1) {
#pragma unroll
      for (int r = 0; r < 8; ++r)
        rm[r] = fmaxf(rm[r], __shfl_xor(rm[r], m, 32));
    }

    float corr[8];
#pragma unroll
    for (int r = 0; r < 8; ++r) {
      const float mn = fmaxf(mrow[r], rm[r]);
      corr[r] = exp2f(mrow[r] - mn);
      mrow[r] = mn;
    }

    float p0[8], p1[8], rs[8];
#pragma unroll
    for (int r = 0; r < 8; ++r) {
      p0[r] = exp2f(sv0[r] - mrow[r]);
      p1[r] = exp2f(sv1[r] - mrow[r]);
      rs[r] = p0[r] + p1[r];
    }
#pragma unroll
    for (int m = 1; m < 16; m <<= 1) {
#pragma unroll
      for (int r = 0; r < 8; ++r)
        rs[r] += __shfl_xor(rs[r], m, 32);
    }
#pragma unroll
    for (int r = 0; r < 8; ++r) lrow[r] = lrow[r] * corr[r] + rs[r];

#pragma unroll
    for (int f = 0; f < 8; ++f)
#pragma unroll
      for (int r = 0; r < 8; ++r) acc[f][r] *= corr[r];

    // ---- P: C-layout -> A-layout via per-wave LDS staging (same-wave, dscnt only) ----
    bf16_t* pw = &Psh[wave * (16 * PSTRIDE)];
#pragma unroll
    for (int r = 0; r < 8; ++r) {
      const int m = r + 8 * hi;
      pw[m * PSTRIDE + ln16]      = (bf16_t)p0[r];
      pw[m * PSTRIDE + 16 + ln16] = (bf16_t)p1[r];
    }
    asm volatile("s_wait_dscnt 0" ::: "memory");
    v16bf pa = lds_load8x2(&pw[ln16 * PSTRIDE + 16 * hi],
                           &pw[ln16 * PSTRIDE + 16 * hi + 8]);

    // ---- O += P V : 8x v_wmma_f32_16x16x32_bf16 over the 128-wide D ----
#pragma unroll
    for (int f = 0; f < 8; ++f) {
      v16bf vb = lds_load16(&VshT[(f * 16 + ln16) * VSTRIDE + 16 * hi]);
      acc[f] = __builtin_amdgcn_wmma_f32_16x16x32_bf16(false, pa, false, vb,
                                                       (short)0, acc[f], false, false);
    }
    __syncthreads();   // protect K/V LDS before next tile's writes
  }

  float inv[8];
#pragma unroll
  for (int r = 0; r < 8; ++r) inv[r] = 1.0f / lrow[r];

#pragma unroll
  for (int f = 0; f < 8; ++f) {
#pragma unroll
    for (int r = 0; r < 8; ++r) {
      const int q = qrow0 + r + 8 * hi;
      Ob[(size_t)q * DHEAD + f * 16 + ln16] = acc[f][r] * inv[r];
    }
  }
}

extern "C" void kernel_launch(void* const* d_in, const int* in_sizes, int n_in,
                              void* d_out, int out_size, void* d_ws, size_t ws_size,
                              hipStream_t stream) {
  const float* q = (const float*)d_in[0];
  const float* k = (const float*)d_in[1];
  const float* v = (const float*)d_in[2];
  float* out = (float*)d_out;

  const int bh = in_sizes[0] / (SEQ_L * DHEAD);   // B*H = 32
  dim3 grid(SEQ_L / BLOCK_M, bh);
  dim3 block(THREADS);
  swa_fwd_kernel<<<grid, block, 0, stream>>>(q, k, v, out);
}
